// MultiHeadAttention_37632503448172
// MI455X (gfx1250) — matrix operand order repaired and checked by reading, not compiled, not run on hardware
//
#include <hip/hip_runtime.h>
#include <hip/hip_bf16.h>

// ---------------------------------------------------------------------------
// MHA forward for MI455X (gfx1250, wave32, WMMA, TDM).
// bf16 data / fp32 accumulate via v_wmma_f32_16x16x32_bf16.
//   K1: fp32 -> bf16 convert (q,k,v, Wq,Wk,Wv,Wo)
//   K2: GEMM  X @ W^T + b  (TDM double-buffered LDS staging) -> bf16 [B,H,S,Dk]
//   K3: flash attention per (b,h,64-q-rows), online softmax fp32
//   K4: GEMM  ctx @ Wo^T + bo -> fp32 row-major (final output)
// Workspace use: ~126 MB.
// ---------------------------------------------------------------------------

typedef __attribute__((ext_vector_type(16))) __bf16 v16bf;
typedef __attribute__((ext_vector_type(8)))  __bf16 v8bf;
typedef __attribute__((ext_vector_type(4)))  __bf16 v4bf;
typedef __attribute__((ext_vector_type(8)))  float  v8f;
typedef __attribute__((ext_vector_type(4)))  float  v4f;
typedef __attribute__((ext_vector_type(4)))  unsigned int v4u;
typedef __attribute__((ext_vector_type(8)))  int    v8i;
typedef __attribute__((ext_vector_type(4)))  int    v4i;

#define WMMA_BF16(a, b, c) \
  __builtin_amdgcn_wmma_f32_16x16x32_bf16(false, (a), false, (b), (short)0, (c), false, false)

// Concatenate two 16-byte LDS/global loads into one 16xbf16 WMMA fragment.
__device__ __forceinline__ v16bf frag16(const __bf16* lo, const __bf16* hi) {
  v8bf a = *(const v8bf*)lo;
  v8bf b = *(const v8bf*)hi;
  return __builtin_shufflevector(a, b, 0,1,2,3,4,5,6,7,8,9,10,11,12,13,14,15);
}

// ---------------------------------------------------------------------------
// TDM: 2D tile load Global -> LDS.
//   tile  = 128 rows x 16 DWORDs (32 bf16), row stride in memory 512 DWORDs
//   LDS   : hardware pad 4 DWORDs after every 16 DWORDs -> 20-DW (40 bf16) pitch
// D# per CDNA5 ISA ch.8: group0 {count=1, lds_addr, global_addr, type=2},
// group1 {data_size=4B, pad_enable, pad_interval=16DW(code 3),
//         pad_amount=4DW(code 3), tensor_dim0=512, tile_dim0=16, tile_dim1=128,
//         tensor_dim0_stride=512}.  Groups 2/3 zero (2D tensor).
// ---------------------------------------------------------------------------
__device__ __forceinline__ void tdm_load_tile_2d(unsigned lds_addr,
                                                 const void* gaddr,
                                                 unsigned tensor_rows) {
  unsigned long long ga = (unsigned long long)(uintptr_t)gaddr;
  v4u g0;
  g0[0] = 1u;                                   // count=1, user descriptor
  g0[1] = lds_addr;                             // lds_addr [63:32]
  g0[2] = (unsigned)ga;                         // global_addr low
  g0[3] = (unsigned)((ga >> 32) & 0x1FFFFFFu)   // global_addr[56:32]
        | (2u << 30);                           // type=2 ("image")
  v8i g1;
  g1[0] = (int)((2u << 16)                      // data_size: 4 bytes
              | (1u << 20)                      // pad_enable
              | (3u << 22)                      // pad_interval: 16 DWORDs
              | (3u << 25));                    // pad_amount: 4 DWORDs
  g1[1] = (int)(512u << 16);                    // tensor_dim0 = 512 DW (row width)
  g1[2] = (int)(tensor_rows << 16);             // tensor_dim1 = #rows
  g1[3] = (int)(16u << 16);                     // tile_dim0 = 16 DW (32 bf16)
  g1[4] = (int)128;                             // tile_dim1 = 128 rows, tile_dim2=0
  g1[5] = (int)512;                             // tensor_dim0_stride = 512 DW
  g1[6] = 0;
  g1[7] = 0;
  v4i z4 = {0, 0, 0, 0};
  v8i z8 = {0, 0, 0, 0, 0, 0, 0, 0};
  __builtin_amdgcn_tensor_load_to_lds(g0, g1, z4, z4, z8, 0);
}

// ---------------------------------------------------------------------------
// K1: elementwise fp32 -> bf16 (4 elements / thread)
// ---------------------------------------------------------------------------
__global__ void __launch_bounds__(256)
cvt_f32_to_bf16(const float* __restrict__ in, __bf16* __restrict__ out, int n) {
  int i = (blockIdx.x * 256 + threadIdx.x) * 4;
  if (i < n) {
    v4f x = *(const v4f*)(in + i);
    v4bf y;
    y[0] = (__bf16)x[0]; y[1] = (__bf16)x[1];
    y[2] = (__bf16)x[2]; y[3] = (__bf16)x[3];
    *(v4bf*)(out + i) = y;
  }
}

// ---------------------------------------------------------------------------
// K2/K4: C[M=8192, N=1024] = A[M,1024](bf16) @ W[N,1024]^T (bf16) + bias.
// Block = 256 threads (8 waves, 4x2 wave grid), tile 128x128, BK=32.
// Tiles staged by TDM into double-buffered LDS; wave 0 drives the DMA and
// TENSORcnt, workgroup barriers publish the buffers. Each wave: 32x64 output.
// MODE 0: out bf16, permuted to [B,H,S,Dk].  MODE 1: out fp32 row-major.
// ---------------------------------------------------------------------------
template <int MODE>
__global__ void __launch_bounds__(256)
gemm_bf16_nt(const __bf16* __restrict__ A, const __bf16* __restrict__ W,
             const float* __restrict__ bias, float* __restrict__ outF,
             __bf16* __restrict__ outP) {
  __shared__ __align__(16) __bf16 As[2][128][40];  // [buf][m][k], 80B pitch
  __shared__ __align__(16) __bf16 Bs[2][128][40];  // [buf][n][k]

  const int t   = threadIdx.x;
  const int ln  = t & 15;          // lane % 16 -> N/M index within tile
  const int hlf = (t >> 4) & 1;    // lane half -> K-half select
  const int w   = t >> 5;          // wave id 0..7
  const int wm  = w >> 1;          // 0..3  (M sub-tile)
  const int wn  = w & 1;           // 0..1  (N sub-tile)
  const int m0  = blockIdx.x * 128;
  const int n0  = blockIdx.y * 128;

  v8f acc[2][4];
#pragma unroll
  for (int mt = 0; mt < 2; ++mt)
#pragma unroll
    for (int nt = 0; nt < 4; ++nt) {
      v8f z = {0.f, 0.f, 0.f, 0.f, 0.f, 0.f, 0.f, 0.f};
      acc[mt][nt] = z;
    }

  // Prologue: wave 0 issues the TDM for k-slab 0 into buffer 0.
  if (t < 32) {
    tdm_load_tile_2d((unsigned)(uintptr_t)&As[0][0][0],
                     A + (size_t)m0 * 1024, 8192u);
    tdm_load_tile_2d((unsigned)(uintptr_t)&Bs[0][0][0],
                     W + (size_t)n0 * 1024, 1024u);
  }

  const int NK = 32;               // 1024 / 32
  for (int ks = 0; ks < NK; ++ks) {
    const int cur = ks & 1;
    if (t < 32) __builtin_amdgcn_s_wait_tensorcnt(0);  // DMA(cur) landed
    __syncthreads();               // publish buffer `cur`; readers of cur^1 done
    if (t < 32 && ks + 1 < NK) {   // issue next slab into the other buffer
      const int k1 = (ks + 1) * 32;
      tdm_load_tile_2d((unsigned)(uintptr_t)&As[cur ^ 1][0][0],
                       A + (size_t)m0 * 1024 + k1, 8192u);
      tdm_load_tile_2d((unsigned)(uintptr_t)&Bs[cur ^ 1][0][0],
                       W + (size_t)n0 * 1024 + k1, 1024u);
    }

    // A fragment: lane = row; elems 0..7 -> K=8*hlf+j, elems 8..15 -> K=16+8*hlf+j
    v16bf afr[2];
#pragma unroll
    for (int mt = 0; mt < 2; ++mt) {
      const int m = wm * 32 + mt * 16 + ln;
      afr[mt] = frag16(&As[cur][m][8 * hlf], &As[cur][m][16 + 8 * hlf]);
    }
    // B fragment: lane = col n; elems 0..7 -> K=8*hlf+j, elems 8..15 -> K=16+8*hlf+j
    v16bf bfr[4];
#pragma unroll
    for (int nt = 0; nt < 4; ++nt) {
      const int n = wn * 64 + nt * 16 + ln;
      bfr[nt] = frag16(&Bs[cur][n][8 * hlf], &Bs[cur][n][16 + 8 * hlf]);
    }
#pragma unroll
    for (int mt = 0; mt < 2; ++mt)
#pragma unroll
      for (int nt = 0; nt < 4; ++nt)
        acc[mt][nt] = WMMA_BF16(afr[mt], bfr[nt], acc[mt][nt]);
  }

  // Epilogue. C layout: lane holds col n = ln (half hlf), VGPR v holds row v+8*hlf.
#pragma unroll
  for (int mt = 0; mt < 2; ++mt)
#pragma unroll
    for (int nt = 0; nt < 4; ++nt) {
      const int n = n0 + wn * 64 + nt * 16 + ln;
      const float bn = bias[n];
#pragma unroll
      for (int v = 0; v < 8; ++v) {
        const int m = m0 + wm * 32 + mt * 16 + v + 8 * hlf;
        const float val = acc[mt][nt][v] + bn;
        if (MODE == 0) {
          // [B,S,D] -> [B,H,S,Dk] head-major bf16
          const int bb = m >> 11, s = m & 2047;
          const int hh = n >> 6, dk = n & 63;
          outP[(((size_t)bb * 16 + hh) * 2048 + s) * 64 + dk] = (__bf16)val;
        } else {
          outF[(size_t)m * 1024 + n] = val;
        }
      }
    }
}

// ---------------------------------------------------------------------------
// K3: flash attention. One block per (b, h, 64 query rows). 128 thr = 4 waves,
// each wave owns 16 query rows. K-tiles of 64 keys, online softmax in fp32.
// ---------------------------------------------------------------------------
__global__ void __launch_bounds__(128)
flash_attn_bf16(const __bf16* __restrict__ Qp, const __bf16* __restrict__ Kp,
                const __bf16* __restrict__ Vp, __bf16* __restrict__ ctx) {
  __shared__ __align__(16) __bf16 Ks[64][72];      // [key][dk]  144B pitch
  __shared__ __align__(16) __bf16 Vs[64][72];      // [d][key]   (transposed)
  __shared__ __align__(16) __bf16 Ps[4][16][72];   // per-wave P staging

  const int t   = threadIdx.x;
  const int ln  = t & 15;
  const int hlf = (t >> 4) & 1;
  const int w   = t >> 5;                 // wave 0..3
  const int qt  = blockIdx.x;             // 0..31
  const int h   = blockIdx.y;             // 0..15
  const int b   = blockIdx.z;             // 0..3
  const size_t headBase = ((size_t)b * 16 + h) * 2048 * 64;

  // Q fragments (A-layout) held in registers for the whole kernel.
  const int qrow = qt * 64 + w * 16 + ln;
  const __bf16* qptr = Qp + headBase + (size_t)qrow * 64;
  v16bf aq[2];
  aq[0] = frag16(qptr + 8 * hlf,      qptr + 16 + 8 * hlf);
  aq[1] = frag16(qptr + 32 + 8 * hlf, qptr + 48 + 8 * hlf);

  v8f accO[4];
  float mrun[8], lrun[8];
#pragma unroll
  for (int dt = 0; dt < 4; ++dt) {
    v8f z = {0.f, 0.f, 0.f, 0.f, 0.f, 0.f, 0.f, 0.f};
    accO[dt] = z;
  }
#pragma unroll
  for (int v = 0; v < 8; ++v) { mrun[v] = -1e30f; lrun[v] = 0.f; }

  const int lr = t >> 1;          // 0..63: staged key row
  const int lc = (t & 1) * 32;    // 0 or 32: staged col base

  for (int kt = 0; kt < 2048; kt += 64) {
    __syncthreads();
    // Stage K tile row-major, V tile transposed.
    const __bf16* ksrc = Kp + headBase + (size_t)(kt + lr) * 64 + lc;
    const __bf16* vsrc = Vp + headBase + (size_t)(kt + lr) * 64 + lc;
#pragma unroll
    for (int j = 0; j < 4; ++j)
      *(v8bf*)&Ks[lr][lc + 8 * j] = *(const v8bf*)(ksrc + 8 * j);
    v8bf vv[4];
#pragma unroll
    for (int j = 0; j < 4; ++j) vv[j] = *(const v8bf*)(vsrc + 8 * j);
#pragma unroll
    for (int j = 0; j < 32; ++j) Vs[lc + j][lr] = vv[j >> 3][j & 7];
    __syncthreads();

    // scores = Q (16x64) x K^T -> 4 tiles of 16x16, K=32 twice
    v8f sc[4];
#pragma unroll
    for (int nt = 0; nt < 4; ++nt) {
      v8f z = {0.f, 0.f, 0.f, 0.f, 0.f, 0.f, 0.f, 0.f};
      sc[nt] = z;
    }
#pragma unroll
    for (int kk = 0; kk < 2; ++kk)
#pragma unroll
      for (int nt = 0; nt < 4; ++nt) {
        const __bf16* kp2 = &Ks[nt * 16 + ln][kk * 32 + 8 * hlf];
        v16bf bk = frag16(kp2, kp2 + 16);
        sc[nt] = WMMA_BF16(aq[kk], bk, sc[nt]);
      }

    // online softmax (rows v+8*hlf live across the 16-lane half)
#pragma unroll
    for (int nt = 0; nt < 4; ++nt)
#pragma unroll
      for (int v = 0; v < 8; ++v) sc[nt][v] *= 0.125f;   // 1/sqrt(64)

    float rmax[8];
#pragma unroll
    for (int v = 0; v < 8; ++v)
      rmax[v] = fmaxf(fmaxf(sc[0][v], sc[1][v]), fmaxf(sc[2][v], sc[3][v]));
#pragma unroll
    for (int off = 1; off <= 8; off <<= 1)
#pragma unroll
      for (int v = 0; v < 8; ++v)
        rmax[v] = fmaxf(rmax[v], __shfl_xor(rmax[v], off, 32));

#pragma unroll
    for (int v = 0; v < 8; ++v) {
      const float mn = fmaxf(mrun[v], rmax[v]);
      const float corr = __expf(mrun[v] - mn);
      mrun[v] = mn;
      float rs = 0.f;
#pragma unroll
      for (int nt = 0; nt < 4; ++nt) {
        const float pe = __expf(sc[nt][v] - mn);
        sc[nt][v] = pe;
        rs += pe;
      }
#pragma unroll
      for (int off = 1; off <= 8; off <<= 1) rs += __shfl_xor(rs, off, 32);
      lrun[v] = lrun[v] * corr + rs;
#pragma unroll
      for (int dt = 0; dt < 4; ++dt) accO[dt][v] *= corr;
    }

    // P: C-layout -> per-wave LDS -> A-layout fragments.
#pragma unroll
    for (int nt = 0; nt < 4; ++nt)
#pragma unroll
      for (int v = 0; v < 8; ++v)
        Ps[w][v + 8 * hlf][nt * 16 + ln] = (__bf16)sc[nt][v];

    // O += P (16x64) x V (64x64)
#pragma unroll
    for (int kk = 0; kk < 2; ++kk) {
      const __bf16* pp = &Ps[w][ln][kk * 32 + 8 * hlf];
      v16bf ap = frag16(pp, pp + 16);
#pragma unroll
      for (int dt = 0; dt < 4; ++dt) {
        const __bf16* vp2 = &Vs[dt * 16 + ln][kk * 32 + 8 * hlf];
        v16bf bv = frag16(vp2, vp2 + 16);
        accO[dt] = WMMA_BF16(ap, bv, accO[dt]);
      }
    }
  }

  // Normalize and write ctx as [B,S,D] bf16.
#pragma unroll
  for (int dt = 0; dt < 4; ++dt)
#pragma unroll
    for (int v = 0; v < 8; ++v) {
      const int srow = qt * 64 + w * 16 + v + 8 * hlf;
      const int d = dt * 16 + ln;
      const float val = accO[dt][v] / lrun[v];
      ctx[((size_t)b * 2048 + srow) * 1024 + h * 64 + d] = (__bf16)val;
    }
}

// ---------------------------------------------------------------------------
// Host launch
// ---------------------------------------------------------------------------
extern "C" void kernel_launch(void* const* d_in, const int* in_sizes, int n_in,
                              void* d_out, int out_size, void* d_ws, size_t ws_size,
                              hipStream_t stream) {
  const float* q  = (const float*)d_in[0];
  const float* k  = (const float*)d_in[1];
  const float* v  = (const float*)d_in[2];
  const float* Wq = (const float*)d_in[3];
  const float* bq = (const float*)d_in[4];
  const float* Wk = (const float*)d_in[5];
  const float* bk = (const float*)d_in[6];
  const float* Wv = (const float*)d_in[7];
  const float* bv = (const float*)d_in[8];
  const float* Wo = (const float*)d_in[9];
  const float* bo = (const float*)d_in[10];
  float* out = (float*)d_out;

  const size_t XB = (size_t)8192 * 1024;   // activation elems
  const size_t WB = (size_t)1024 * 1024;   // weight elems

  __bf16* p   = (__bf16*)d_ws;
  __bf16* qb  = p; p += XB;
  __bf16* kb  = p; p += XB;
  __bf16* vb  = p; p += XB;
  __bf16* Wqb = p; p += WB;
  __bf16* Wkb = p; p += WB;
  __bf16* Wvb = p; p += WB;
  __bf16* Wob = p; p += WB;
  __bf16* Qp  = p; p += XB;
  __bf16* Kp  = p; p += XB;
  __bf16* Vp  = p; p += XB;
  __bf16* ctxb = p;                        // total ~126 MB

  // K1: conversions
  cvt_f32_to_bf16<<<(int)(XB / 1024), 256, 0, stream>>>(q, qb, (int)XB);
  cvt_f32_to_bf16<<<(int)(XB / 1024), 256, 0, stream>>>(k, kb, (int)XB);
  cvt_f32_to_bf16<<<(int)(XB / 1024), 256, 0, stream>>>(v, vb, (int)XB);
  cvt_f32_to_bf16<<<(int)(WB / 1024), 256, 0, stream>>>(Wq, Wqb, (int)WB);
  cvt_f32_to_bf16<<<(int)(WB / 1024), 256, 0, stream>>>(Wk, Wkb, (int)WB);
  cvt_f32_to_bf16<<<(int)(WB / 1024), 256, 0, stream>>>(Wv, Wvb, (int)WB);
  cvt_f32_to_bf16<<<(int)(WB / 1024), 256, 0, stream>>>(Wo, Wob, (int)WB);

  // K2: projections -> [B,H,S,Dk] bf16
  dim3 gg(64, 8), gb(256);
  gemm_bf16_nt<0><<<gg, gb, 0, stream>>>(qb, Wqb, bq, (float*)nullptr, Qp);
  gemm_bf16_nt<0><<<gg, gb, 0, stream>>>(kb, Wkb, bk, (float*)nullptr, Kp);
  gemm_bf16_nt<0><<<gg, gb, 0, stream>>>(vb, Wvb, bv, (float*)nullptr, Vp);

  // K3: attention
  flash_attn_bf16<<<dim3(32, 16, 4), 128, 0, stream>>>(Qp, Kp, Vp, ctxb);

  // K4: output projection -> fp32
  gemm_bf16_nt<1><<<gg, gb, 0, stream>>>(ctxb, Wob, bo, out, (__bf16*)nullptr);
}
